// MPNN_60593398612224
// MI455X (gfx1250) — matrix operand order repaired and checked by reading, not compiled, not run on hardware
//
#include <hip/hip_runtime.h>
#include <hip/hip_bf16.h>

#define H      128
#define DIN    256
#define DE     64
#define KP1    352   // MSG_IN = 323 padded to 11*32
#define NLAYER 3
#define NGRAPH 64

typedef __attribute__((ext_vector_type(16))) __bf16 v16bf;
typedef __attribute__((ext_vector_type(8)))  __bf16 v8bf;
typedef __attribute__((ext_vector_type(8)))  float  v8f;

union AB16 { v16bf v; v8bf h[2]; };

__device__ __forceinline__ unsigned short f2bf(float f) {
  union { float f; unsigned u; } v; v.f = f;
  unsigned r = v.u + 0x7FFFu + ((v.u >> 16) & 1u);   // round-to-nearest-even
  return (unsigned short)(r >> 16);
}

__device__ __forceinline__ v8f wmma_bf16(v16bf a, v16bf b, v8f c) {
  // (neg_a, A, neg_b, B, c_mod, C, reuse_a, reuse_b)
  return __builtin_amdgcn_wmma_f32_16x16x32_bf16(false, a, false, b, (short)0, c,
                                                 false, false);
}

// Async Global -> LDS copy of 16 bytes (per lane), tracked by ASYNCcnt.
__device__ __forceinline__ void async_g2l_b128(unsigned lds_off, const void* gptr) {
  asm volatile("global_load_async_to_lds_b128 %0, %1, off"
               :: "v"(lds_off), "v"((unsigned long long)(size_t)gptr)
               : "memory");
}
__device__ __forceinline__ void wait_asynccnt0() {
  asm volatile("s_wait_asynccnt 0" ::: "memory");
}

// ---------------------------------------------------------------------------
// Weight prep: fp32 [K][Nc] -> bf16 transposed+padded [Nc][Kpad]
// ---------------------------------------------------------------------------
__global__ __launch_bounds__(256) void k_prep_w(const float* __restrict__ src,
                                                unsigned short* __restrict__ dst,
                                                int K, int Nc, int Kpad) {
  int i = blockIdx.x * 256 + threadIdx.x;
  if (i >= Nc * Kpad) return;
  int n = i / Kpad, k = i % Kpad;
  dst[i] = (k < K) ? f2bf(src[(size_t)k * Nc + n]) : (unsigned short)0;
}

// ---------------------------------------------------------------------------
// h = x @ Wi + bi  (N x 256 x 128), stores f32 and bf16 copies of h
// ---------------------------------------------------------------------------
__global__ __launch_bounds__(128) void k_input_linear(
    const float* __restrict__ x, const unsigned short* __restrict__ Wt,  // [H][DIN]
    const float* __restrict__ bias, float* __restrict__ hf,
    unsigned short* __restrict__ hbf, int N) {
  const int tid = threadIdx.x;
  const int wave = tid >> 5, lane = tid & 31;
  const int l15 = lane & 15, hi = lane >> 4;
  const int base = blockIdx.x * 64 + wave * 16;
  int rowA = base + l15; if (rowA > N - 1) rowA = N - 1;

  v8f acc[8];
#pragma unroll
  for (int nt = 0; nt < 8; ++nt) {
    float bb = bias[nt * 16 + l15];
#pragma unroll
    for (int r = 0; r < 8; ++r) acc[nt][r] = bb;
  }
#pragma unroll
  for (int kc = 0; kc < DIN / 32; ++kc) {
    const float* ap = x + (size_t)rowA * DIN + kc * 32 + hi * 16;
    float4 f0 = *(const float4*)(ap);
    float4 f1 = *(const float4*)(ap + 4);
    float4 f2 = *(const float4*)(ap + 8);
    float4 f3 = *(const float4*)(ap + 12);
    union { v16bf v; unsigned short u[16]; } a;
    a.u[0] = f2bf(f0.x);  a.u[1] = f2bf(f0.y);  a.u[2] = f2bf(f0.z);  a.u[3] = f2bf(f0.w);
    a.u[4] = f2bf(f1.x);  a.u[5] = f2bf(f1.y);  a.u[6] = f2bf(f1.z);  a.u[7] = f2bf(f1.w);
    a.u[8] = f2bf(f2.x);  a.u[9] = f2bf(f2.y);  a.u[10] = f2bf(f2.z); a.u[11] = f2bf(f2.w);
    a.u[12] = f2bf(f3.x); a.u[13] = f2bf(f3.y); a.u[14] = f2bf(f3.z); a.u[15] = f2bf(f3.w);
    v16bf bfrag[8];
#pragma unroll
    for (int nt = 0; nt < 8; ++nt)
      bfrag[nt] = *(const v16bf*)(Wt + (size_t)(nt * 16 + l15) * DIN + kc * 32 + hi * 16);
#pragma unroll
    for (int nt = 0; nt < 8; ++nt) acc[nt] = wmma_bf16(a.v, bfrag[nt], acc[nt]);
  }
#pragma unroll
  for (int nt = 0; nt < 8; ++nt) {
    int n = nt * 16 + l15;
#pragma unroll
    for (int r = 0; r < 8; ++r) {
      int row = base + hi * 8 + r;
      if (row < N) {
        float y = acc[nt][r];
        hf[(size_t)row * H + n] = y;
        hbf[(size_t)row * H + n] = f2bf(y);
      }
    }
  }
}

// ---------------------------------------------------------------------------
// Fused message pass: async gather -> MLP1 -> BN -> ReLU -> MLP2 -> scatter-add
// 64 edges per block, 4 waves; each wave owns a 16-row M tile.
// ---------------------------------------------------------------------------
__global__ __launch_bounds__(128) void k_message(
    const unsigned short* __restrict__ hbf, const int* __restrict__ ei,
    const float* __restrict__ eattr, const float* __restrict__ pos,
    const unsigned short* __restrict__ W1t,  // [H][KP1]
    const float* __restrict__ b1, const float* __restrict__ g1,
    const float* __restrict__ be1,
    const unsigned short* __restrict__ W2t,  // [H][H]
    const float* __restrict__ b2, float* __restrict__ aggr, int N, int E) {
  __shared__ unsigned short Abuf[64 * KP1];  // 45056 B
  __shared__ unsigned short Mbuf[64 * H];    // 16384 B
  const int total = E + N;
  const int tid = threadIdx.x;
  const int tbase = blockIdx.x * 64;

  // ---- stage 1: build bf16 A tile [xi | xj | ea | rel | pad] ----
  {
    int r = tid >> 1, part = tid & 1;
    int e = tbase + r;
    int sidx = 0, didx = 0;
    bool valid = (e < total);
    bool realE = valid && (e < E);
    if (realE)      { sidx = ei[e]; didx = ei[E + e]; }
    else if (valid) { sidx = didx = e - E; }
    unsigned short* rowp = Abuf + r * KP1;
    if (valid) {
      // async DMA of the two gathered h rows straight into LDS (no VGPR bounce)
      unsigned ldi = (unsigned)(size_t)(void*)(rowp + part * 64);
      unsigned ldj = (unsigned)(size_t)(void*)(rowp + H + part * 64);
      const unsigned short* xi = hbf + (size_t)didx * H + part * 64;
      const unsigned short* xj = hbf + (size_t)sidx * H + part * 64;
#pragma unroll
      for (int k = 0; k < 8; ++k) {
        async_g2l_b128(ldi + 16u * k, xi + 8 * k);
        async_g2l_b128(ldj + 16u * k, xj + 8 * k);
      }
    } else {
      uint4 z; z.x = z.y = z.z = z.w = 0u;
      uint4* di = (uint4*)(rowp + part * 64);
      uint4* dj = (uint4*)(rowp + H + part * 64);
#pragma unroll
      for (int k = 0; k < 8; ++k) { di[k] = z; dj[k] = z; }
    }
    unsigned short* dea = rowp + 2 * H + part * 32;
    if (realE) {
      const float* ap = eattr + (size_t)e * DE + part * 32;
#pragma unroll
      for (int j = 0; j < 32; ++j) dea[j] = f2bf(ap[j]);
    } else {
#pragma unroll
      for (int j = 0; j < 32; ++j) dea[j] = 0;
    }
    if (part == 0) {
      float r0 = 0.f, r1 = 0.f, r2 = 0.f;
      if (realE) {
        r0 = pos[(size_t)didx * 3 + 0] - pos[(size_t)sidx * 3 + 0];
        r1 = pos[(size_t)didx * 3 + 1] - pos[(size_t)sidx * 3 + 1];
        r2 = pos[(size_t)didx * 3 + 2] - pos[(size_t)sidx * 3 + 2];
      }
      rowp[320] = f2bf(r0); rowp[321] = f2bf(r1); rowp[322] = f2bf(r2);
      for (int c = 323; c < 336; ++c) rowp[c] = 0;
    } else {
      for (int c = 336; c < KP1; ++c) rowp[c] = 0;
    }
  }
  wait_asynccnt0();     // async Global->LDS transfers complete
  __syncthreads();

  const int wave = tid >> 5, lane = tid & 31;
  const int l15 = lane & 15, hi = lane >> 4;
  const int mrow = wave * 16;
  const float bnr = rsqrtf(1.0f + 1e-5f);

  // ---- stage 2: m1 = relu(bn(A @ W1 + b1)) -> Mbuf (bf16) ----
  {
    v8f acc[8];
#pragma unroll
    for (int nt = 0; nt < 8; ++nt) {
      float bb = b1[nt * 16 + l15];
#pragma unroll
      for (int r = 0; r < 8; ++r) acc[nt][r] = bb;
    }
#pragma unroll
    for (int kc = 0; kc < KP1 / 32; ++kc) {
      AB16 a;
      const unsigned short* ar = Abuf + (mrow + l15) * KP1 + kc * 32 + hi * 16;
      a.h[0] = *(const v8bf*)ar;
      a.h[1] = *(const v8bf*)(ar + 8);
      if (kc + 1 < KP1 / 32)
        __builtin_prefetch(W1t + (size_t)l15 * KP1 + (kc + 1) * 32 + hi * 16, 0, 0);
      v16bf bfrag[8];
#pragma unroll
      for (int nt = 0; nt < 8; ++nt)
        bfrag[nt] = *(const v16bf*)(W1t + (size_t)(nt * 16 + l15) * KP1 + kc * 32 + hi * 16);
#pragma unroll
      for (int nt = 0; nt < 8; ++nt) acc[nt] = wmma_bf16(a.v, bfrag[nt], acc[nt]);
    }
#pragma unroll
    for (int nt = 0; nt < 8; ++nt) {
      int n = nt * 16 + l15;
      float sc = g1[n] * bnr;
      float sh = be1[n];
#pragma unroll
      for (int r = 0; r < 8; ++r) {
        float z = fmaxf(acc[nt][r] * sc + sh, 0.f);
        Mbuf[(mrow + hi * 8 + r) * H + n] = f2bf(z);
      }
    }
  }
  __syncthreads();

  // ---- stage 3: m2 = m1 @ W2 + b2, scatter-add into aggr[dst] ----
  {
    v8f acc[8];
#pragma unroll
    for (int nt = 0; nt < 8; ++nt) {
      float bb = b2[nt * 16 + l15];
#pragma unroll
      for (int r = 0; r < 8; ++r) acc[nt][r] = bb;
    }
#pragma unroll
    for (int kc = 0; kc < H / 32; ++kc) {
      AB16 a;
      const unsigned short* ar = Mbuf + (mrow + l15) * H + kc * 32 + hi * 16;
      a.h[0] = *(const v8bf*)ar;
      a.h[1] = *(const v8bf*)(ar + 8);
      v16bf bfrag[8];
#pragma unroll
      for (int nt = 0; nt < 8; ++nt)
        bfrag[nt] = *(const v16bf*)(W2t + (size_t)(nt * 16 + l15) * H + kc * 32 + hi * 16);
#pragma unroll
      for (int nt = 0; nt < 8; ++nt) acc[nt] = wmma_bf16(a.v, bfrag[nt], acc[nt]);
    }
#pragma unroll
    for (int r = 0; r < 8; ++r) {
      int row = mrow + hi * 8 + r;
      int e = tbase + row;
      if (e < total) {
        int didx = (e < E) ? ei[E + e] : (e - E);
        float* dp = aggr + (size_t)didx * H + l15;
#pragma unroll
        for (int nt = 0; nt < 8; ++nt) atomicAdd(dp + nt * 16, acc[nt][r]);
      }
    }
  }
}

// ---------------------------------------------------------------------------
// h += relu(h @ nodeW + b + aggr); refresh bf16 copy
// ---------------------------------------------------------------------------
__global__ __launch_bounds__(128) void k_node_update(
    unsigned short* __restrict__ hbf, const unsigned short* __restrict__ Wt,  // [H][H]
    const float* __restrict__ bias, const float* __restrict__ aggr,
    float* __restrict__ hf, int N) {
  const int tid = threadIdx.x;
  const int wave = tid >> 5, lane = tid & 31;
  const int l15 = lane & 15, hi = lane >> 4;
  const int base = blockIdx.x * 64 + wave * 16;
  int rowA = base + l15; if (rowA > N - 1) rowA = N - 1;

  v8f acc[8];
#pragma unroll
  for (int nt = 0; nt < 8; ++nt) {
    float bb = bias[nt * 16 + l15];
#pragma unroll
    for (int r = 0; r < 8; ++r) acc[nt][r] = bb;
  }
#pragma unroll
  for (int kc = 0; kc < H / 32; ++kc) {
    AB16 a;
    const unsigned short* ar = hbf + (size_t)rowA * H + kc * 32 + hi * 16;
    a.h[0] = *(const v8bf*)ar;
    a.h[1] = *(const v8bf*)(ar + 8);
    v16bf bfrag[8];
#pragma unroll
    for (int nt = 0; nt < 8; ++nt)
      bfrag[nt] = *(const v16bf*)(Wt + (size_t)(nt * 16 + l15) * H + kc * 32 + hi * 16);
#pragma unroll
    for (int nt = 0; nt < 8; ++nt) acc[nt] = wmma_bf16(a.v, bfrag[nt], acc[nt]);
  }
#pragma unroll
  for (int nt = 0; nt < 8; ++nt) {
    int n = nt * 16 + l15;
#pragma unroll
    for (int r = 0; r < 8; ++r) {
      int row = base + hi * 8 + r;
      if (row < N) {
        float u = fmaxf(acc[nt][r] + aggr[(size_t)row * H + n], 0.f);
        float hv = hf[(size_t)row * H + n] + u;
        hf[(size_t)row * H + n] = hv;
        hbf[(size_t)row * H + n] = f2bf(hv);
      }
    }
  }
}

// ---------------------------------------------------------------------------
// Segment sums + counts (atomics)
// ---------------------------------------------------------------------------
__global__ __launch_bounds__(256) void k_pool(const float* __restrict__ hf,
                                              const int* __restrict__ batch,
                                              float* __restrict__ sums,
                                              float* __restrict__ cnts, int N) {
  int gid = blockIdx.x * 256 + threadIdx.x;
  int node = gid >> 5;
  if (node >= N) return;
  int f = (gid & 31) * 4;
  int b = batch[node];
  const float* hp = hf + (size_t)node * H + f;
  float* sp = sums + (size_t)b * H + f;
  atomicAdd(sp + 0, hp[0]);
  atomicAdd(sp + 1, hp[1]);
  atomicAdd(sp + 2, hp[2]);
  atomicAdd(sp + 3, hp[3]);
  if ((gid & 31) == 0) atomicAdd(cnts + b, 1.0f);
}

// ---------------------------------------------------------------------------
// Tiny output head: pooled -> linear/BN/ReLU -> linear
// ---------------------------------------------------------------------------
__global__ __launch_bounds__(128) void k_head(
    const float* __restrict__ sums, const float* __restrict__ cnts,
    const float* __restrict__ W1, const float* __restrict__ b1,
    const float* __restrict__ g, const float* __restrict__ be,
    const float* __restrict__ W2, const float* __restrict__ b2,
    float* __restrict__ out) {
  __shared__ float P[NGRAPH * H];
  __shared__ float T[NGRAPH * H];
  int tid = threadIdx.x;
  for (int i = tid; i < NGRAPH * H; i += 128) {
    int gph = i / H;
    P[i] = sums[i] / fmaxf(cnts[gph], 1.0f);
  }
  __syncthreads();
  const float bnr = rsqrtf(1.0f + 1e-5f);
  int n = tid;
  float sc = g[n] * bnr, sh = be[n];
  for (int gph = 0; gph < NGRAPH; ++gph) {
    float s = b1[n];
    for (int k = 0; k < H; ++k) s += P[gph * H + k] * W1[k * H + n];
    T[gph * H + n] = fmaxf(s * sc + sh, 0.f);
  }
  __syncthreads();
  for (int gph = 0; gph < NGRAPH; ++gph) {
    float s = b2[n];
    for (int k = 0; k < H; ++k) s += T[gph * H + k] * W2[k * H + n];
    out[gph * H + n] = s;
  }
}

// ---------------------------------------------------------------------------
extern "C" void kernel_launch(void* const* d_in, const int* in_sizes, int n_in,
                              void* d_out, int out_size, void* d_ws, size_t ws_size,
                              hipStream_t stream) {
  const float* x     = (const float*)d_in[0];
  const int*   ei    = (const int*)d_in[1];
  const float* eattr = (const float*)d_in[2];
  const float* pos   = (const float*)d_in[3];
  const int*   batch = (const int*)d_in[4];
  const float* Wi    = (const float*)d_in[5];
  const float* bi    = (const float*)d_in[6];
  const float* nodeW = (const float*)d_in[7];
  const float* nodeB = (const float*)d_in[8];
  const float* mW1   = (const float*)d_in[9];
  const float* mB1   = (const float*)d_in[10];
  const float* mG    = (const float*)d_in[11];
  const float* mBe   = (const float*)d_in[12];
  const float* mW2   = (const float*)d_in[13];
  const float* mB2   = (const float*)d_in[14];
  const float* oW1   = (const float*)d_in[15];
  const float* oB1   = (const float*)d_in[16];
  const float* oG    = (const float*)d_in[17];
  const float* oBe   = (const float*)d_in[18];
  const float* oW2   = (const float*)d_in[19];
  const float* oB2   = (const float*)d_in[20];

  const int N = in_sizes[0] / DIN;
  const int E = in_sizes[1] / 2;

  char* ws = (char*)d_ws;
  size_t off = 0;
  auto walloc = [&](size_t bytes) -> void* {
    void* p = ws + off;
    off += (bytes + 255) & ~(size_t)255;
    return p;
  };
  float*          hf   = (float*)walloc((size_t)N * H * 4);
  unsigned short* hbf  = (unsigned short*)walloc((size_t)N * H * 2);
  float*          aggr = (float*)walloc((size_t)N * H * 4);
  unsigned short* Wit  = (unsigned short*)walloc((size_t)H * DIN * 2);
  unsigned short* nWt  = (unsigned short*)walloc((size_t)NLAYER * H * H * 2);
  unsigned short* mW1t = (unsigned short*)walloc((size_t)NLAYER * H * KP1 * 2);
  unsigned short* mW2t = (unsigned short*)walloc((size_t)NLAYER * H * H * 2);
  float*          sums = (float*)walloc((size_t)NGRAPH * H * 4);
  float*          cnts = (float*)walloc((size_t)NGRAPH * 4);
  (void)ws_size; (void)n_in; (void)out_size;

  // weight prep (bf16, transposed, padded)
  k_prep_w<<<(H * DIN + 255) / 256, 256, 0, stream>>>(Wi, Wit, DIN, H, DIN);
  for (int l = 0; l < NLAYER; ++l) {
    k_prep_w<<<(H * H + 255) / 256, 256, 0, stream>>>(
        nodeW + (size_t)l * H * H, nWt + (size_t)l * H * H, H, H, H);
    k_prep_w<<<(H * KP1 + 255) / 256, 256, 0, stream>>>(
        mW1 + (size_t)l * 323 * H, mW1t + (size_t)l * H * KP1, 323, H, KP1);
    k_prep_w<<<(H * H + 255) / 256, 256, 0, stream>>>(
        mW2 + (size_t)l * H * H, mW2t + (size_t)l * H * H, H, H, H);
  }

  k_input_linear<<<(N + 63) / 64, 128, 0, stream>>>(x, Wit, bi, hf, hbf, N);

  const int total = E + N;
  for (int l = 0; l < NLAYER; ++l) {
    hipMemsetAsync(aggr, 0, (size_t)N * H * 4, stream);
    k_message<<<(total + 63) / 64, 128, 0, stream>>>(
        hbf, ei, eattr, pos,
        mW1t + (size_t)l * H * KP1, mB1 + (size_t)l * H, mG + (size_t)l * H,
        mBe + (size_t)l * H, mW2t + (size_t)l * H * H, mB2 + (size_t)l * H,
        aggr, N, E);
    k_node_update<<<(N + 63) / 64, 128, 0, stream>>>(
        hbf, nWt + (size_t)l * H * H, nodeB + (size_t)l * H, aggr, hf, N);
  }

  hipMemsetAsync(sums, 0, (size_t)NGRAPH * H * 4, stream);
  hipMemsetAsync(cnts, 0, (size_t)NGRAPH * 4, stream);
  k_pool<<<((unsigned)N * 32 + 255) / 256, 256, 0, stream>>>(hf, batch, sums, cnts, N);
  k_head<<<1, 128, 0, stream>>>(sums, cnts, oW1, oB1, oG, oBe, oW2, oB2,
                                (float*)d_out);
}
